// TrellisLinear_62079457296919
// MI455X (gfx1250) — compile-verified, matrix operand order checked by reading, not run on hardware
//
#include <hip/hip_runtime.h>
#include <stdint.h>

#define M_DIM 512
#define K_DIM 8192
#define N_DIM 8192
#define GROUPK 128
#define BM 128
#define BN 128
#define KB 64
#define KBP 72   // padded LDS row length in halves: 144B = 16B-aligned, bank-conflict-free
#define NCHUNK (K_DIM / KB)

typedef __attribute__((ext_vector_type(16))) __bf16 v16bf;
typedef __attribute__((ext_vector_type(8)))  float  v8f;

union FragAB { uint4 q[2]; v16bf v; };

// fp32 -> bf16 (RNE) pair pack: 2x round + 1x v_perm_b32 merge
static __device__ __forceinline__ unsigned int pack_bf16x2(float a, float b) {
  union { float f; unsigned int u; } ua, ub;
  ua.f = a; ub.f = b;
  unsigned int ra = ua.u + 0x7FFFu + ((ua.u >> 16) & 1u);
  unsigned int rb = ub.u + 0x7FFFu + ((ub.u >> 16) & 1u);
  return __builtin_amdgcn_perm(rb, ra, 0x07060302);  // {rb[31:16], ra[31:16]}
}

static __device__ __forceinline__ int i4comp(const int4& p, int i) {
  return (i == 0) ? p.x : (i == 1) ? p.y : (i == 2) ? p.z : p.w;
}

__global__ __launch_bounds__(256)
void trellis_wmma_kernel(const float* __restrict__ x,
                         const int* __restrict__ packed,
                         const float* __restrict__ grid,
                         const float* __restrict__ scales,
                         const float* __restrict__ su,
                         const float* __restrict__ sv,
                         float* __restrict__ out) {
  __shared__ unsigned short As[2][BM][KBP];   // x tile, bf16, row-major over K (double-buffered)
  __shared__ unsigned short Bs[2][BN][KBP];   // W tile, bf16, transposed n-major over K
  __shared__ float gridS[16];

  const int tid  = threadIdx.x;
  const int lane = tid & 31;
  const int wid  = tid >> 5;
  const int wm   = wid & 1;     // 64-row slab
  const int wn   = wid >> 1;    // 32-col slab

  const int m0 = blockIdx.x * BM;   // M fast axis -> concurrent blocks share packed stripe in L2
  const int n0 = blockIdx.y * BN;

  // x staging: thread -> (row, 32-col half)
  const int xr = tid >> 1;
  const int xh = tid & 1;
  // dequant: thread -> 4 byte-pairs (8 n-cols) x 4 consecutive k-rows
  const int pg = tid & 15;
  const int kb = tid >> 4;

  if (tid < 16) gridS[tid] = grid[tid];

  // -------- register-staged chunk (software pipeline) --------
  float4 xq[8];
  int4   pq[4];
  float  sc8[8], sv8[8], su4[4];

  auto load_chunk = [&](int k0) {
    const float* xp = x + (size_t)(m0 + xr) * K_DIM + k0 + xh * 32;
#pragma unroll
    for (int q = 0; q < 8; ++q) xq[q] = ((const float4*)xp)[q];
#pragma unroll
    for (int r = 0; r < 4; ++r)
      pq[r] = *(const int4*)&packed[(size_t)(k0 + kb * 4 + r) * (N_DIM / 2) + (n0 >> 1) + pg * 4];
    const int kg = k0 / GROUPK;   // KB=64 never straddles a 128-row scale group
#pragma unroll
    for (int jj = 0; jj < 8; ++jj) {
      const int n = n0 + pg * 8 + jj;
      sc8[jj] = scales[(size_t)kg * N_DIM + n];
      sv8[jj] = sv[n];
    }
#pragma unroll
    for (int r = 0; r < 4; ++r) su4[r] = su[k0 + kb * 4 + r];
    if (k0 + KB < K_DIM)
      __builtin_prefetch(&packed[(size_t)(k0 + KB + kb * 4) * (N_DIM / 2) + (n0 >> 1) + pg * 4], 0, 3);
  };

  auto store_chunk = [&](int buf) {
    // x tile fp32 -> bf16, ds_store_b128
#pragma unroll
    for (int q = 0; q < 4; ++q) {
      uint4 pk;
      pk.x = pack_bf16x2(xq[2 * q].x,     xq[2 * q].y);
      pk.y = pack_bf16x2(xq[2 * q].z,     xq[2 * q].w);
      pk.z = pack_bf16x2(xq[2 * q + 1].x, xq[2 * q + 1].y);
      pk.w = pack_bf16x2(xq[2 * q + 1].z, xq[2 * q + 1].w);
      *(uint4*)&As[buf][xr][xh * 32 + q * 8] = pk;
    }
    // W dequant: 4 consecutive k per n-column -> one ds_store_b64 per column
#pragma unroll
    for (int ii = 0; ii < 4; ++ii) {
      const int b0 = i4comp(pq[0], ii);
      const int b1 = i4comp(pq[1], ii);
      const int b2 = i4comp(pq[2], ii);
      const int b3 = i4comp(pq[3], ii);
#pragma unroll
      for (int h = 0; h < 2; ++h) {
        const int   nl = pg * 8 + ii * 2 + h;
        const float s  = sc8[ii * 2 + h] * sv8[ii * 2 + h];
        const float w0 = gridS[(b0 >> (4 * h)) & 15] * s * su4[0];
        const float w1 = gridS[(b1 >> (4 * h)) & 15] * s * su4[1];
        const float w2 = gridS[(b2 >> (4 * h)) & 15] * s * su4[2];
        const float w3 = gridS[(b3 >> (4 * h)) & 15] * s * su4[3];
        uint2 st;
        st.x = pack_bf16x2(w0, w1);
        st.y = pack_bf16x2(w2, w3);
        *(uint2*)&Bs[buf][nl][kb * 4] = st;   // 8B-aligned ds_store_b64
      }
    }
  };

  v8f acc[4][2];
#pragma unroll
  for (int i = 0; i < 4; ++i)
#pragma unroll
    for (int j = 0; j < 2; ++j)
      acc[i][j] = (v8f){};

  const int loff = (lane >= 16) ? 8 : 0;   // 16-bit A/B layout: hi-lanes hold K+8
  const int l16  = lane & 15;

  auto mma_chunk = [&](int buf) {
#pragma unroll
    for (int ks = 0; ks < KB; ks += 32) {
      FragAB bfr[2];
#pragma unroll
      for (int j = 0; j < 2; ++j) {
        const unsigned short* bp = &Bs[buf][wn * 32 + j * 16 + l16][ks + loff];
        bfr[j].q[0] = *(const uint4*)(bp);
        bfr[j].q[1] = *(const uint4*)(bp + 16);
      }
#pragma unroll
      for (int i = 0; i < 4; ++i) {
        FragAB afr;
        const unsigned short* ap = &As[buf][wm * 64 + i * 16 + l16][ks + loff];
        afr.q[0] = *(const uint4*)(ap);
        afr.q[1] = *(const uint4*)(ap + 16);
#pragma unroll
        for (int j = 0; j < 2; ++j) {
          acc[i][j] = __builtin_amdgcn_wmma_f32_16x16x32_bf16(
              false, afr.v, false, bfr[j].v, (short)0, acc[i][j], false, false);
        }
      }
    }
  };

  // -------- pipelined main loop (branch-free steady state) --------
  // prologue
  load_chunk(0);
  __syncthreads();          // gridS visible before first dequant
  store_chunk(0);
  __syncthreads();

  // steady state: load(k+1) ; WMMA(k) ; convert/store(k+1) ; barrier
  for (int kc = 0; kc < NCHUNK - 1; ++kc) {
    const int cur = kc & 1;
    load_chunk((kc + 1) * KB);   // issue global loads into regs (no waits here)
    mma_chunk(cur);              // WMMA hides the load latency
    store_chunk(cur ^ 1);        // first use of loaded regs -> waits sink here
    __syncthreads();             // single barrier per iteration
  }
  // epilogue: last chunk, MMA only
  mma_chunk((NCHUNK - 1) & 1);

  // -------- writeback: C layout = VGPR v -> row (v | v+8 by lane half), col = lane%16 --------
#pragma unroll
  for (int i = 0; i < 4; ++i) {
    const int rbase = m0 + wm * 64 + i * 16 + ((lane >= 16) ? 8 : 0);
#pragma unroll
    for (int j = 0; j < 2; ++j) {
      const int c = n0 + wn * 32 + j * 16 + (lane & 15);
#pragma unroll
      for (int v = 0; v < 8; ++v) {
        out[(size_t)(rbase + v) * N_DIM + c] = acc[i][j][v];
      }
    }
  }
}

extern "C" void kernel_launch(void* const* d_in, const int* in_sizes, int n_in,
                              void* d_out, int out_size, void* d_ws, size_t ws_size,
                              hipStream_t stream) {
  (void)in_sizes; (void)n_in; (void)out_size; (void)d_ws; (void)ws_size;
  const float* x      = (const float*)d_in[0];
  const int*   packed = (const int*)d_in[1];
  const float* grid   = (const float*)d_in[2];
  const float* scales = (const float*)d_in[3];
  const float* su     = (const float*)d_in[4];
  const float* sv     = (const float*)d_in[5];
  float* out = (float*)d_out;

  dim3 g(M_DIM / BM, N_DIM / BN);   // (4, 64); M fast for packed-stripe L2 reuse
  trellis_wmma_kernel<<<g, dim3(256), 0, stream>>>(x, packed, grid, scales, su, sv, out);
}
